// NeuroMemory_67456756351028
// MI455X (gfx1250) — compile-verified, matrix operand order checked by reading, not run on hardware
//
#include <hip/hip_runtime.h>

// ---------------------------------------------------------------------------
// Problem constants (from the reference)
// ---------------------------------------------------------------------------
static constexpr int kB   = 32;
static constexpr int kS   = 256;
static constexpr int kHid = 1024;
static constexpr int kMem = 256;
static constexpr int kNW  = 1024;

// ---------------------------------------------------------------------------
// WMMA types / helpers (CDNA5 gfx1250, wave32)
// ---------------------------------------------------------------------------
typedef __attribute__((ext_vector_type(16))) __bf16 v16bf;
typedef __attribute__((ext_vector_type(8)))  __bf16 v8bf;
typedef __attribute__((ext_vector_type(4)))  __bf16 v4bf;
typedef __attribute__((ext_vector_type(8)))  float  v8f;

// Load a 16-element bf16 fragment (A or B operand) for one lane.
// Per ISA 16-bit A/B layout: lane covers row (lane%16); VGPRs 0..3 hold
// K = half*8 .. +7 and VGPRs 4..7 hold K = 16+half*8 .. +7 (two contiguous
// 16-byte runs -> two ds_load_b128 per fragment).
static __device__ __forceinline__ v16bf ld_frag(const __bf16* __restrict__ base, int half) {
  v8bf lo = *(const v8bf*)(base + half * 8);
  v8bf hi = *(const v8bf*)(base + 16 + half * 8);
  v16bf r;
#pragma unroll
  for (int i = 0; i < 8; ++i) { r[i] = lo[i]; r[8 + i] = hi[i]; }
  return r;
}

static __device__ __forceinline__ v8f wmma_bf16(v16bf a, v16bf b, v8f c) {
  // 8 args: (neg_a, A, neg_b, B, c_mod, C, reuse_a, reuse_b)
  return __builtin_amdgcn_wmma_f32_16x16x32_bf16(false, a, false, b, (short)0, c, false, false);
}

// ---------------------------------------------------------------------------
// Tiled GEMM:  C[M,N] = A[M,K](lda) * B[N,K](ldb)^T  (+bias) (+=C)
// fp32 in/out, bf16 WMMA math, fp32 accumulation.
// Block: 256 threads (8 waves), tile 128(M) x 128(N), K-step 32.
// Wave grid 4x2: each wave computes 32x64 (2 A-frags x 4 B-frags = 8 WMMA).
// Register double-buffering: next global tile is loaded while WMMAs run.
// Requires: K % 32 == 0, N % 128 == 0. EDGE covers ragged M (M=10 bank).
// ---------------------------------------------------------------------------
template <bool EDGE>
__global__ __launch_bounds__(256) void gemm_abT_kernel(
    const float* __restrict__ A, int lda,
    const float* __restrict__ Bm, int ldb,
    float* __restrict__ C, int ldc,
    const float* __restrict__ bias,
    int M, int N, int K, int betaFlag) {
  constexpr int KP = 40;  // 32 + 8 bf16 pad (80B row stride, 16B aligned)
  __shared__ __attribute__((aligned(16))) __bf16 As[128 * KP];
  __shared__ __attribute__((aligned(16))) __bf16 Bs[128 * KP];

  const int tid  = threadIdx.x;
  const int wid  = tid >> 5;
  const int lane = tid & 31;
  const int lr   = lane & 15;
  const int half = lane >> 4;
  const int m0   = blockIdx.y * 128;
  const int n0   = blockIdx.x * 128;
  const int wy   = wid >> 1;  // 0..3: rows wy*32 .. wy*32+31
  const int wx   = wid & 1;   // 0..1: cols wx*64 .. wx*64+63

  float4 ra[4], rb[4];

  auto load_regs = [&](int k0) {
#pragma unroll
    for (int i = 0; i < 4; ++i) {
      const int f = i * 256 + tid;
      const int r = f >> 3, c4 = (f & 7) * 4;
      if constexpr (EDGE) {
        ra[i] = make_float4(0.f, 0.f, 0.f, 0.f);
        if (m0 + r < M) ra[i] = *(const float4*)(A + (size_t)(m0 + r) * lda + k0 + c4);
      } else {
        ra[i] = *(const float4*)(A + (size_t)(m0 + r) * lda + k0 + c4);
      }
      rb[i] = *(const float4*)(Bm + (size_t)(n0 + r) * ldb + k0 + c4);
    }
  };

  auto store_regs = [&]() {
#pragma unroll
    for (int i = 0; i < 4; ++i) {
      const int f = i * 256 + tid;
      const int r = f >> 3, c4 = (f & 7) * 4;
      v4bf pa; pa[0] = (__bf16)ra[i].x; pa[1] = (__bf16)ra[i].y;
               pa[2] = (__bf16)ra[i].z; pa[3] = (__bf16)ra[i].w;
      *(v4bf*)&As[r * KP + c4] = pa;
      v4bf pb; pb[0] = (__bf16)rb[i].x; pb[1] = (__bf16)rb[i].y;
               pb[2] = (__bf16)rb[i].z; pb[3] = (__bf16)rb[i].w;
      *(v4bf*)&Bs[r * KP + c4] = pb;
    }
  };

  v8f acc[2][4] = {};

  load_regs(0);
  for (int k0 = 0; k0 < K; k0 += 32) {
    store_regs();
    __syncthreads();
    if (k0 + 32 < K) load_regs(k0 + 32);  // overlap next global tile with WMMAs
    if (k0 + 64 < K) {                    // light L2 prefetch two slabs ahead
      __builtin_prefetch(A  + (size_t)(m0 + (tid & 127)) * lda + k0 + 64, 0, 1);
      __builtin_prefetch(Bm + (size_t)(n0 + (tid & 127)) * ldb + k0 + 64, 0, 1);
    }

    // Gather all fragments first, then burst the 8 WMMAs.
    v16bf af0 = ld_frag(&As[(wy * 32 + lr) * KP], half);
    v16bf af1 = ld_frag(&As[(wy * 32 + 16 + lr) * KP], half);
    v16bf bfr[4];
#pragma unroll
    for (int j = 0; j < 4; ++j)
      bfr[j] = ld_frag(&Bs[(wx * 64 + j * 16 + lr) * KP], half);
#pragma unroll
    for (int j = 0; j < 4; ++j) {
      acc[0][j] = wmma_bf16(af0, bfr[j], acc[0][j]);
      acc[1][j] = wmma_bf16(af1, bfr[j], acc[1][j]);
    }
    __syncthreads();
  }

  // Epilogue: C layout — lane covers column (lane%16); VGPR r is row half*8+r.
#pragma unroll
  for (int mt = 0; mt < 2; ++mt) {
#pragma unroll
    for (int j = 0; j < 4; ++j) {
      const int col = n0 + wx * 64 + j * 16 + lr;
      const float bv = bias ? bias[col] : 0.0f;
#pragma unroll
      for (int r = 0; r < 8; ++r) {
        const int row = m0 + wy * 32 + mt * 16 + half * 8 + r;
        if (!EDGE || row < M) {
          float v = acc[mt][j][r] + bv;
          if (betaFlag) v += C[(size_t)row * ldc + col];
          C[(size_t)row * ldc + col] = v;
        }
      }
    }
  }
}

// ---------------------------------------------------------------------------
// Fused attention over a shared memory bank:
//   scores = Qh * Kh^T * scale ; softmax over M ; ctx = P * Vh
// One block per (S-chunk of 128 rows, head, batch). 8 waves x 16 rows each.
// K row-major bf16, V transposed bf16 (both WMMA B-operands read contiguous
// K-runs). Probabilities round-trip through LDS to re-fragment as A-operand.
// ---------------------------------------------------------------------------
template <int D, int MM>
constexpr unsigned attn_lds_bytes() {
  constexpr int MC = ((MM + 31) / 32) * 32;
  constexpr int KP = D + 8;
  constexpr int MP = MC + 8;
  return 2u * (128 * KP + MC * KP + D * MP + 128 * MP);
}

template <int D, int MM>
__global__ __launch_bounds__(256) void attn_kernel(
    const float* __restrict__ qh, const float* __restrict__ kh,
    const float* __restrict__ vh, float* __restrict__ ctx, float scale) {
  constexpr int MC    = ((MM + 31) / 32) * 32;  // K-dim padded to 32
  constexpr int NT    = MC / 16;                // score N-tiles
  constexpr int ND    = D / 16;                 // output N-tiles
  constexpr int KP    = D + 8;
  constexpr int MP    = MC + 8;
  constexpr int SROWS = 128;
  constexpr int GS    = (NT < 4) ? NT : 4;      // fragment batching
  constexpr int GO    = (ND < 4) ? ND : 4;

  extern __shared__ char dynsmem[];
  __bf16* Qs  = (__bf16*)dynsmem;        // [SROWS][KP]
  __bf16* Ks  = Qs + SROWS * KP;         // [MC][KP]   (zero-padded rows)
  __bf16* Vst = Ks + MC * KP;            // [D][MP]    V transposed
  __bf16* Att = Vst + D * MP;            // [SROWS][MP]

  const int tid   = threadIdx.x;
  const int chunk = blockIdx.x;
  const int h     = blockIdx.y;
  const int b     = blockIdx.z;

  const size_t qrow0 = (size_t)b * kS + (size_t)chunk * SROWS;
  const float* qbase = qh + qrow0 * kHid + h * D;
  const float* kbase = kh + h * D;
  const float* vbase = vh + h * D;

  for (int idx = tid; idx < SROWS * D; idx += 256) {
    int r = idx / D, c = idx % D;
    Qs[r * KP + c] = (__bf16)qbase[(size_t)r * kHid + c];
  }
  for (int idx = tid; idx < MC * D; idx += 256) {
    int r = idx / D, c = idx % D;
    float kv = (r < MM) ? kbase[(size_t)r * kHid + c] : 0.0f;
    float vv = (r < MM) ? vbase[(size_t)r * kHid + c] : 0.0f;
    Ks[r * KP + c]  = (__bf16)kv;
    Vst[c * MP + r] = (__bf16)vv;
  }
  __syncthreads();

  const int wid  = tid >> 5;
  const int lane = tid & 31;
  const int lr   = lane & 15;
  const int half = lane >> 4;
  const int mrow = wid * 16;

  // ---- scores: [16 x MC] strip per wave ----
  v8f sacc[NT] = {};
  const __bf16* aQ = &Qs[(mrow + lr) * KP];
#pragma unroll
  for (int kk = 0; kk < D; kk += 32) {
    v16bf a = ld_frag(aQ + kk, half);
#pragma unroll
    for (int t0 = 0; t0 < NT; t0 += GS) {
      v16bf bfr[GS];
#pragma unroll
      for (int j = 0; j < GS; ++j)
        bfr[j] = ld_frag(&Ks[((t0 + j) * 16 + lr) * KP + kk], half);
#pragma unroll
      for (int j = 0; j < GS; ++j)
        sacc[t0 + j] = wmma_bf16(a, bfr[j], sacc[t0 + j]);
    }
  }

  // ---- softmax over M (row spans 16 lanes of the same half) ----
  const float NEG_INF = -__builtin_inff();
  float rm[8];
#pragma unroll
  for (int r = 0; r < 8; ++r) rm[r] = NEG_INF;
#pragma unroll
  for (int t = 0; t < NT; ++t) {
    const bool ok = (t * 16 + lr) < MM;
#pragma unroll
    for (int r = 0; r < 8; ++r) {
      float v = ok ? sacc[t][r] * scale : NEG_INF;
      sacc[t][r] = v;
      rm[r] = fmaxf(rm[r], v);
    }
  }
#pragma unroll
  for (int off = 1; off < 16; off <<= 1)
#pragma unroll
    for (int r = 0; r < 8; ++r)
      rm[r] = fmaxf(rm[r], __shfl_xor(rm[r], off, 32));

  float sm[8];
#pragma unroll
  for (int r = 0; r < 8; ++r) sm[r] = 0.0f;
#pragma unroll
  for (int t = 0; t < NT; ++t) {
#pragma unroll
    for (int r = 0; r < 8; ++r) {
      float e = __expf(sacc[t][r] - rm[r]);  // masked -> exp(-inf) = 0
      sacc[t][r] = e;
      sm[r] += e;
    }
  }
#pragma unroll
  for (int off = 1; off < 16; off <<= 1)
#pragma unroll
    for (int r = 0; r < 8; ++r)
      sm[r] += __shfl_xor(sm[r], off, 32);
#pragma unroll
  for (int r = 0; r < 8; ++r) sm[r] = 1.0f / sm[r];

  // Probabilities -> this wave's private Att strip (same-wave LDS is in-order).
#pragma unroll
  for (int t = 0; t < NT; ++t) {
    int col = t * 16 + lr;
#pragma unroll
    for (int r = 0; r < 8; ++r)
      Att[(mrow + half * 8 + r) * MP + col] = (__bf16)(sacc[t][r] * sm[r]);
  }

  // ---- ctx = P * V ----
  v8f oacc[ND] = {};
  const __bf16* aA = &Att[(mrow + lr) * MP];
#pragma unroll
  for (int kk = 0; kk < MC; kk += 32) {
    v16bf a = ld_frag(aA + kk, half);
#pragma unroll
    for (int t0 = 0; t0 < ND; t0 += GO) {
      v16bf bfr[GO];
#pragma unroll
      for (int j = 0; j < GO; ++j)
        bfr[j] = ld_frag(&Vst[((t0 + j) * 16 + lr) * MP + kk], half);
#pragma unroll
      for (int j = 0; j < GO; ++j)
        oacc[t0 + j] = wmma_bf16(a, bfr[j], oacc[t0 + j]);
    }
  }

#pragma unroll
  for (int t = 0; t < ND; ++t) {
    int col = h * D + t * 16 + lr;
#pragma unroll
    for (int r = 0; r < 8; ++r) {
      size_t row = qrow0 + mrow + half * 8 + r;
      ctx[row * kHid + col] = oacc[t][r];
    }
  }
}

// ---------------------------------------------------------------------------
// Episodic-write helper kernels
// ---------------------------------------------------------------------------
__global__ __launch_bounds__(256) void softmax_rows_kernel(float* __restrict__ p, float mul) {
  __shared__ float red[256];
  const int row = blockIdx.x, t = threadIdx.x;
  float v = p[(size_t)row * kMem + t];
  red[t] = v;
  __syncthreads();
  for (int s = 128; s > 0; s >>= 1) { if (t < s) red[t] = fmaxf(red[t], red[t + s]); __syncthreads(); }
  float mx = red[0];
  __syncthreads();
  float e = __expf(v - mx);
  red[t] = e;
  __syncthreads();
  for (int s = 128; s > 0; s >>= 1) { if (t < s) red[t] += red[t + s]; __syncthreads(); }
  p[(size_t)row * kMem + t] = e * (mul / red[0]);
}

__global__ void colmean_kernel(const float* __restrict__ p, float* __restrict__ u, int rows) {
  const int m = threadIdx.x;
  float s = 0.f;
  for (int n = 0; n < rows; ++n) s += p[(size_t)n * kMem + m];
  u[m] = s / (float)rows;
}

// writes[m,h] = (1/NW) * sum_n probs[n,m]*data[n,h];  ek2/ev2 = bank*(1-u)+writes
// Each thread accumulates 16 m-slots per data element read -> data is read
// only kMem/16 = 16 times total (~64 MB) instead of 256 times (~1 GB).
__global__ __launch_bounds__(256) void memwrite_kernel(
    const float* __restrict__ probs, const float* __restrict__ data,
    const float* __restrict__ u, const float* __restrict__ ek,
    const float* __restrict__ ev, float* __restrict__ ek2, float* __restrict__ ev2) {
  const int h  = blockIdx.x * 256 + threadIdx.x;
  const int m0 = blockIdx.y * 16;
  float s[16];
#pragma unroll
  for (int j = 0; j < 16; ++j) s[j] = 0.f;
  for (int n = 0; n < kNW; ++n) {
    const float d = data[(size_t)n * kHid + h];          // coalesced
    const float* pr = probs + (size_t)n * kMem + m0;     // uniform across block
#pragma unroll
    for (int j = 0; j < 16; ++j) s[j] = fmaf(pr[j], d, s[j]);
  }
#pragma unroll
  for (int j = 0; j < 16; ++j) {
    const int m = m0 + j;
    const float wv = s[j] * (1.0f / kNW);
    const float decay = 1.0f - u[m];
    const size_t idx = (size_t)m * kHid + h;
    ek2[idx] = ek[idx] * decay + wv;
    ev2[idx] = ev[idx] * decay + wv;
  }
}

// ---------------------------------------------------------------------------
// Host-side launch
// ---------------------------------------------------------------------------
static inline void launch_gemm(const float* A, int lda, const float* B, int ldb,
                               float* C, int ldc, const float* bias,
                               int M, int N, int K, int beta, hipStream_t s) {
  dim3 g(N / 128, (M + 127) / 128);
  if (M % 128 == 0)
    gemm_abT_kernel<false><<<g, 256, 0, s>>>(A, lda, B, ldb, C, ldc, bias, M, N, K, beta);
  else
    gemm_abT_kernel<true><<<g, 256, 0, s>>>(A, lda, B, ldb, C, ldc, bias, M, N, K, beta);
}

extern "C" void kernel_launch(void* const* d_in, const int* in_sizes, int n_in,
                              void* d_out, int out_size, void* d_ws, size_t ws_size,
                              hipStream_t stream) {
  const float* q    = (const float*)d_in[0];
  const float* data = (const float*)d_in[1];
  const float* epk  = (const float*)d_in[2];
  const float* epv  = (const float*)d_in[3];
  const float* smk  = (const float*)d_in[4];
  const float* smv  = (const float*)d_in[5];
  const float* wm   = (const float*)d_in[6];
  const float* in_w[3]  = {(const float*)d_in[7],  (const float*)d_in[11], (const float*)d_in[15]};
  const float* in_b[3]  = {(const float*)d_in[8],  (const float*)d_in[12], (const float*)d_in[16]};
  const float* out_w[3] = {(const float*)d_in[9],  (const float*)d_in[13], (const float*)d_in[17]};
  const float* out_b[3] = {(const float*)d_in[10], (const float*)d_in[14], (const float*)d_in[18]};
  const float* proj_w = (const float*)d_in[19];
  const float* proj_b = (const float*)d_in[20];
  float* out = (float*)d_out;

  // Workspace partition (floats); total ~18.1M floats (~72 MB).
  float* ws    = (float*)d_ws;
  float* probs = ws;                    ws += (size_t)kNW * kMem;
  float* u     = ws;                    ws += kMem;
  float* ek2   = ws;                    ws += (size_t)kMem * kHid;
  float* ev2   = ws;                    ws += (size_t)kMem * kHid;
  float* khb   = ws;                    ws += (size_t)kMem * kHid;
  float* vhb   = ws;                    ws += (size_t)kMem * kHid;
  float* qhb   = ws;                    ws += (size_t)kB * kS * kHid;
  float* ctxb  = ws;

  const int R = kB * kS;  // 8192 query rows

  // ---- episodic fast-weight write ----
  launch_gemm(data, kHid, epk, kHid, probs, kMem, nullptr, kNW, kMem, kHid, 0, stream);
  softmax_rows_kernel<<<kNW, kMem, 0, stream>>>(probs, 0.1f);  // PLAST*IMP
  colmean_kernel<<<1, kMem, 0, stream>>>(probs, u, kNW);
  memwrite_kernel<<<dim3(kHid / 256, kMem / 16), 256, 0, stream>>>(
      probs, data, u, epk, epv, ek2, ev2);

  // ---- three attentions, out-proj, and folded final projection ----
  const float* km[3] = {ek2, smk, wm};
  const float* vm[3] = {ev2, smv, wm};
  const int    Mm[3] = {kMem, kMem, 10};

  for (int c = 0; c < 3; ++c) {
    // qh = q @ wq^T + bq ; kh/vh from memory bank
    launch_gemm(q, kHid, in_w[c], kHid, qhb, kHid, in_b[c], R, kHid, kHid, 0, stream);
    launch_gemm(km[c], kHid, in_w[c] + (size_t)kHid * kHid, kHid, khb, kHid,
                in_b[c] + kHid, Mm[c], kHid, kHid, 0, stream);
    launch_gemm(vm[c], kHid, in_w[c] + 2 * (size_t)kHid * kHid, kHid, vhb, kHid,
                in_b[c] + 2 * kHid, Mm[c], kHid, kHid, 0, stream);

    if (c < 2) {
      attn_kernel<64, 256><<<dim3(kS / 128, 16, kB), 256, attn_lds_bytes<64, 256>(), stream>>>(
          qhb, khb, vhb, ctxb, 0.125f);                      // 1/sqrt(64)
    } else {
      attn_kernel<128, 10><<<dim3(kS / 128, 8, kB), 256, attn_lds_bytes<128, 10>(), stream>>>(
          qhb, khb, vhb, ctxb, 0.088388347648318447f);      // 1/sqrt(128)
    }

    // r = ctx @ out_w^T + out_b   (reuse qhb — qh dead after attention)
    launch_gemm(ctxb, kHid, out_w[c], kHid, qhb, kHid, out_b[c], R, kHid, kHid, 0, stream);
    // out (+)= r @ proj_w[:, c*H:(c+1)*H]^T   (+ proj_b on first chunk)
    launch_gemm(qhb, kHid, proj_w + (size_t)c * kHid, 3 * kHid, out, kHid,
                (c == 0) ? proj_b : nullptr, R, kHid, kHid, (c == 0) ? 0 : 1, stream);
  }
}
